// DeformableAttention_43619687858465
// MI455X (gfx1250) — matrix operand order repaired and checked by reading, not compiled, not run on hardware
//
#include <hip/hip_runtime.h>
#include <math.h>

// ---------------------------------------------------------------------------
// Deformable attention for MI455X (gfx1250, wave32).
// - Projections: bf16 WMMA GEMMs (v_wmma_f32_16x16x32_bf16, fp32 accumulate)
// - GEMM tiles staged to LDS by the Tensor Data Mover (tensor_load_to_lds,
//   TENSORcnt double-buffered) when the builtin is available; plain b128
//   staging otherwise.
// - Bilinear gather phase is VALU/L2-bound (proj = 32MB, L2-resident in 192MB)
// ---------------------------------------------------------------------------

typedef __bf16 bf16_t;
typedef __attribute__((ext_vector_type(16))) __bf16    v16bf;
typedef __attribute__((ext_vector_type(8)))  float     v8f;
typedef __attribute__((ext_vector_type(4)))  unsigned  v4u;
typedef __attribute__((ext_vector_type(8)))  int       v8i;
typedef __attribute__((ext_vector_type(4)))  int       v4i;

#define NHEAD 8
#define NPTS  4
#define BATCH 8
#define NQ    4096
#define CDIM  256
#define HD    32
#define HSZ   64
#define MROWS (BATCH * NQ)          // 32768

#define BM 128
#define BN 64
#define BK 32
#define LSTR 40   // LDS row stride in bf16 (80B): 16B-aligned, conflict-free reads

#define USE_TDM __has_builtin(__builtin_amdgcn_tensor_load_to_lds)

static __device__ __forceinline__ unsigned short bfbits(float f) {
    unsigned u = __builtin_bit_cast(unsigned, f);
    unsigned r = u + 0x7FFFu + ((u >> 16) & 1u);       // RNE
    return (unsigned short)(r >> 16);
}
static __device__ __forceinline__ bf16_t f2bf(float f) {
    unsigned short s = bfbits(f);
    return __builtin_bit_cast(bf16_t, s);
}

#if USE_TDM
// 2D tile DMA: global (row-major, rows of `tensor_d0` elems, 2B each) -> LDS
// at byte offset lds_off, rows padded 64B -> 80B (pad 4 DWORDs per 16 DWORDs).
static __device__ __forceinline__ void tdm_load_2d(unsigned lds_off,
                                                   const bf16_t* gptr,
                                                   int tensor_d0, int tensor_d1,
                                                   int tile_d0, int tile_d1) {
    unsigned long long ga = (unsigned long long)gptr;
    v4u g0;
    g0[0] = 1u;                                         // count=1, load, user mode
    g0[1] = lds_off;                                    // LDS byte address
    g0[2] = (unsigned)(ga & 0xFFFFFFFFu);               // global_addr[31:0]
    g0[3] = (unsigned)((ga >> 32) & 0x1FFFFFFu) | (2u << 30);  // addr[56:32], type=2
    v8i g1;
    g1[0] = (int)((1u << 16)      // data_size = 2 bytes
                | (1u << 20)      // pad_enable
                | (3u << 22)      // pad_interval: 16 DWORDs (64B) per pad
                | (3u << 25));    // pad_amount: 4 DWORDs (16B)
    g1[1] = (int)(((unsigned)tensor_d0 & 0xFFFFu) << 16);               // dim0[15:0]
    g1[2] = (int)(((unsigned)tensor_d0 >> 16) | ((unsigned)tensor_d1 << 16));
    g1[3] = (int)(((unsigned)tensor_d1 >> 16) | ((unsigned)tile_d0 << 16));
    g1[4] = (int)((unsigned)tile_d1 & 0xFFFFu);                          // tile_dim1
    g1[5] = tensor_d0;                                                   // dim0_stride lo
    g1[6] = 0;
    g1[7] = 0;
    v4i z4 = {0, 0, 0, 0};
#if defined(__clang_major__) && (__clang_major__ >= 23)
    v8i z8 = {0, 0, 0, 0, 0, 0, 0, 0};
    __builtin_amdgcn_tensor_load_to_lds(g0, g1, z4, z4, z8, 0);
#else
    __builtin_amdgcn_tensor_load_to_lds(g0, g1, z4, z4, 0);
#endif
}
static __device__ __forceinline__ unsigned lds_off_of(const void* p) {
    // generic pointer to LDS carries the LDS byte offset in its low 32 bits
    return (unsigned)(unsigned long long)p;
}
#endif

// ---------------------------------------------------------------------------
// Out[M,N](f32) = A[M,K](bf16) * Wt[N,K](bf16, pre-transposed) + bias[N]
// BM=128, BN=64, BK=32; 8 waves, 4(M)x2(N), each wave = 2x2 16x16 fragments.
// ---------------------------------------------------------------------------
__global__ __launch_bounds__(256)
void gemm_bf16_wmma(const bf16_t* __restrict__ A, const bf16_t* __restrict__ Wt,
                    const float* __restrict__ bias, float* __restrict__ Out,
                    int M, int N, int K) {
    __shared__ bf16_t As[2][BM * LSTR];   // [m][k] rows at 80B stride
    __shared__ bf16_t Bs[2][BN * LSTR];   // [n][k] rows at 80B stride

    const int tid   = threadIdx.x;
    const int lane  = tid & 31;
    const int wav   = tid >> 5;
    const int waveM = wav >> 1, waveN = wav & 1;
    const int bm0   = blockIdx.y * BM;
    const int bn0   = blockIdx.x * BN;

    v8f acc[2][2] = {};
    const int T = K / BK;

#if USE_TDM
    if (wav == 0) {
        tdm_load_2d(lds_off_of(&As[0][0]), A + (size_t)bm0 * K, K, M, BK, BM);
        tdm_load_2d(lds_off_of(&Bs[0][0]), Wt + (size_t)bn0 * K, K, N, BK, BN);
    }
#endif

    for (int i = 0; i < T; ++i) {
        const int buf = i & 1;
#if USE_TDM
        if (wav == 0) {
            if (i + 1 < T) {
                const int kb2 = (i + 1) * BK;
                tdm_load_2d(lds_off_of(&As[buf ^ 1][0]), A + (size_t)bm0 * K + kb2, K, M, BK, BM);
                tdm_load_2d(lds_off_of(&Bs[buf ^ 1][0]), Wt + (size_t)bn0 * K + kb2, K, N, BK, BN);
                __builtin_amdgcn_s_wait_tensorcnt((short)2);  // tile i done, i+1 in flight
            } else {
                __builtin_amdgcn_s_wait_tensorcnt((short)0);
            }
        }
        __syncthreads();
#else
        // fallback: stage bf16 tiles with b128 loads/stores (single layout match)
        {
            const int kb = i * BK;
            #pragma unroll
            for (int s = 0; s < 2; ++s) {               // A: 512 x 16B segments
                const int seg = tid * 2 + s;
                const int r = seg >> 2, q = seg & 3;
                const uint4 d = *(const uint4*)&A[(size_t)(bm0 + r) * K + kb + q * 8];
                *(uint4*)&As[buf][r * LSTR + q * 8] = d;
            }
            {                                            // B: 256 x 16B segments
                const int r = tid >> 2, q = tid & 3;
                const uint4 d = *(const uint4*)&Wt[(size_t)(bn0 + r) * K + kb + q * 8];
                *(uint4*)&Bs[buf][r * LSTR + q * 8] = d;
            }
        }
        __syncthreads();
#endif
        union Frag { v16bf v; unsigned u[8]; };
        Frag a[2], b[2];
        const int nlane  = lane & 15;
        const int aKBase = (lane < 16) ? 0 : 16;   // 16-bit A 16x32 layout (contiguous k, matches B)
        const int bKBase = (lane < 16) ? 0 : 16;   // 16-bit B 32x16 layout
        #pragma unroll
        for (int mi = 0; mi < 2; ++mi) {
            const int row = waveM * 32 + mi * 16 + nlane;
            #pragma unroll
            for (int j = 0; j < 8; ++j) {
                const int kk = aKBase + 2 * j;
                a[mi].u[j] = *(const unsigned*)&As[buf][row * LSTR + kk];
            }
        }
        #pragma unroll
        for (int ni = 0; ni < 2; ++ni) {
            const int col = waveN * 32 + ni * 16 + nlane;
            #pragma unroll
            for (int j = 0; j < 8; ++j)
                b[ni].u[j] = *(const unsigned*)&Bs[buf][col * LSTR + bKBase + 2 * j];
        }
        #pragma unroll
        for (int mi = 0; mi < 2; ++mi)
            #pragma unroll
            for (int ni = 0; ni < 2; ++ni)
                acc[mi][ni] = __builtin_amdgcn_wmma_f32_16x16x32_bf16(
                    false, a[mi].v, false, b[ni].v, (short)0, acc[mi][ni],
                    false, false);
        __syncthreads();
    }

    // epilogue: C/D layout: VGPR r -> M = r + (lane>=16 ? 8 : 0), N = lane&15
    const int nlane = lane & 15;
    const int mHi   = (lane >= 16) ? 8 : 0;
    #pragma unroll
    for (int ni = 0; ni < 2; ++ni) {
        const int n  = bn0 + waveN * 32 + ni * 16 + nlane;
        const float bv = bias[n];
        #pragma unroll
        for (int mi = 0; mi < 2; ++mi) {
            #pragma unroll
            for (int r = 0; r < 8; ++r) {
                const int m = bm0 + waveM * 32 + mi * 16 + mHi + r;
                Out[(size_t)m * N + n] = acc[mi][ni][r] + bv;
            }
        }
    }
}

// ---------------------------------------------------------------------------
// fp32 -> bf16 streaming convert (4 elems/thread)
// ---------------------------------------------------------------------------
__global__ __launch_bounds__(256)
void cvt_bf16(const float* __restrict__ in, bf16_t* __restrict__ out, int n4) {
    const int i = blockIdx.x * 256 + threadIdx.x;
    if (i < n4) {
        const float4 f = ((const float4*)in)[i];
        ushort4 o;
        o.x = bfbits(f.x); o.y = bfbits(f.y); o.z = bfbits(f.z); o.w = bfbits(f.w);
        ((ushort4*)out)[i] = o;
    }
}

// W[K][N] fp32 -> Wt[N][K] bf16 (weights are tiny; done once per launch)
__global__ __launch_bounds__(256)
void wt_transpose(const float* __restrict__ W, bf16_t* __restrict__ Wt, int K, int N) {
    const int g = blockIdx.x * 256 + threadIdx.x;
    if (g < K * N) {
        const int n = g / K, k = g - n * K;
        Wt[g] = f2bf(W[(size_t)k * N + n]);
    }
}

// [W_off | W_attn | pad] -> Wt_cat[128][256] bf16 (transposed) + bcat[128]
__global__ __launch_bounds__(256)
void prep_cat_t(const float* __restrict__ W_off, const float* __restrict__ b_off,
                const float* __restrict__ W_attn, const float* __restrict__ b_attn,
                bf16_t* __restrict__ Wt_cat, float* __restrict__ bcat) {
    const int g = blockIdx.x * 256 + threadIdx.x;
    if (g < 128 * 256) {
        const int n = g >> 8, k = g & 255;
        float v = 0.0f;
        if (n < 64)      v = W_off[k * 64 + n];
        else if (n < 96) v = W_attn[k * 32 + (n - 64)];
        Wt_cat[g] = f2bf(v);
    }
    if (g < 128) {
        float v = 0.0f;
        if (g < 64)      v = b_off[g];
        else if (g < 96) v = b_attn[g - 64];
        bcat[g] = v;
    }
}

// ---------------------------------------------------------------------------
// Bilinear sampling + softmax-weighted sum; one wave per (b, n, head), lane =
// channel. Emits bf16 so the output projection GEMM consumes it directly.
// ---------------------------------------------------------------------------
__global__ __launch_bounds__(256)
void deform_sample(const float* __restrict__ proj,     // [MROWS][256] fp32
                   const float* __restrict__ offattn,  // [MROWS][128] fp32
                   const float* __restrict__ ref,      // [MROWS][2]
                   bf16_t* __restrict__ mix) {         // [MROWS][256] bf16
    const int lane = threadIdx.x & 31;
    const int wav  = (int)((blockIdx.x * blockDim.x + threadIdx.x) >> 5);
    const int h    = wav & (NHEAD - 1);
    const int row  = wav >> 3;             // b*4096 + n
    const int b    = row >> 12;

    const float rx = ref[row * 2 + 0];
    const float ry = ref[row * 2 + 1];
    const float* oa = offattn + (size_t)row * 128;

    float lg[NPTS];
    #pragma unroll
    for (int p = 0; p < NPTS; ++p) lg[p] = oa[64 + h * 4 + p];
    const float mx = fmaxf(fmaxf(lg[0], lg[1]), fmaxf(lg[2], lg[3]));
    float es = 0.0f;
    #pragma unroll
    for (int p = 0; p < NPTS; ++p) { lg[p] = __expf(lg[p] - mx); es += lg[p]; }
    const float inv = 1.0f / es;

    const float* pbase = proj + ((size_t)b * (HSZ * HSZ)) * CDIM + h * HD + lane;
    float acc = 0.0f;
    #pragma unroll
    for (int p = 0; p < NPTS; ++p) {
        const float ap = lg[p] * inv;
        const float ox = tanhf(oa[h * 8 + p * 2 + 0]) * 0.5f;
        const float oy = tanhf(oa[h * 8 + p * 2 + 1]) * 0.5f;
        const float x = fminf(fmaxf(rx + ox, 0.0f), 1.0f) * (float)HSZ - 0.5f;
        const float y = fminf(fmaxf(ry + oy, 0.0f), 1.0f) * (float)HSZ - 0.5f;
        const float x0f = floorf(x), y0f = floorf(y);
        const float wx1 = x - x0f, wx0 = 1.0f - wx1;
        const float wy1 = y - y0f, wy0 = 1.0f - wy1;
        const int ix0 = (int)x0f, ix1 = ix0 + 1;
        const int iy0 = (int)y0f, iy1 = iy0 + 1;
        if (ix0 >= 0 && iy0 >= 0)
            acc += ap * wx0 * wy0 * pbase[(size_t)(iy0 * HSZ + ix0) * CDIM];
        if (ix1 < HSZ && iy0 >= 0)
            acc += ap * wx1 * wy0 * pbase[(size_t)(iy0 * HSZ + ix1) * CDIM];
        if (ix0 >= 0 && iy1 < HSZ)
            acc += ap * wx0 * wy1 * pbase[(size_t)(iy1 * HSZ + ix0) * CDIM];
        if (ix1 < HSZ && iy1 < HSZ)
            acc += ap * wx1 * wy1 * pbase[(size_t)(iy1 * HSZ + ix1) * CDIM];
    }
    mix[(size_t)row * CDIM + h * HD + lane] = f2bf(acc);
}

// ---------------------------------------------------------------------------
extern "C" void kernel_launch(void* const* d_in, const int* in_sizes, int n_in,
                              void* d_out, int out_size, void* d_ws, size_t ws_size,
                              hipStream_t stream) {
    const float* query  = (const float*)d_in[0];
    const float* value  = (const float*)d_in[1];
    const float* ref    = (const float*)d_in[2];
    const float* W_off  = (const float*)d_in[3];
    const float* b_off  = (const float*)d_in[4];
    const float* W_attn = (const float*)d_in[5];
    const float* b_attn = (const float*)d_in[6];
    const float* W_val  = (const float*)d_in[7];
    const float* b_val  = (const float*)d_in[8];
    const float* W_out  = (const float*)d_in[9];
    const float* b_out  = (const float*)d_in[10];
    float* out = (float*)d_out;
    (void)in_sizes; (void)n_in; (void)out_size; (void)ws_size; // needs ~97 MB ws

    char* ws = (char*)d_ws;
    bf16_t* Wt_val  = (bf16_t*)(ws + 0);                         // 128 KB
    bf16_t* Wt_out  = (bf16_t*)(ws + (128u << 10));              // 128 KB
    bf16_t* Wt_cat  = (bf16_t*)(ws + (256u << 10));              // 64 KB
    float*  bcat    = (float*)(ws + (320u << 10));               // 512 B
    bf16_t* val_bf  = (bf16_t*)(ws + (1u << 20));                // 16 MB
    bf16_t* qry_bf  = (bf16_t*)(ws + (17u << 20));               // 16 MB
    bf16_t* mix_bf  = (bf16_t*)(ws + (33u << 20));               // 16 MB
    float*  proj    = (float*)(ws + (49u << 20));                // 32 MB
    float*  offatt  = (float*)(ws + (81u << 20));                // 16 MB

    const int NELEM = MROWS * CDIM;          // 8388608
    cvt_bf16<<<(NELEM / 4 + 255) / 256, 256, 0, stream>>>(value, val_bf, NELEM / 4);
    cvt_bf16<<<(NELEM / 4 + 255) / 256, 256, 0, stream>>>(query, qry_bf, NELEM / 4);
    wt_transpose<<<(256 * 256 + 255) / 256, 256, 0, stream>>>(W_val, Wt_val, 256, 256);
    wt_transpose<<<(256 * 256 + 255) / 256, 256, 0, stream>>>(W_out, Wt_out, 256, 256);
    prep_cat_t<<<128, 256, 0, stream>>>(W_off, b_off, W_attn, b_attn, Wt_cat, bcat);

    // value projection: [32768,256] x [256,256]
    gemm_bf16_wmma<<<dim3(256 / BN, MROWS / BM), 256, 0, stream>>>(
        val_bf, Wt_val, b_val, proj, MROWS, 256, 256);

    // fused offset+attention logits: [32768,256] x [256,128]
    gemm_bf16_wmma<<<dim3(128 / BN, MROWS / BM), 256, 0, stream>>>(
        qry_bf, Wt_cat, bcat, offatt, MROWS, 128, 256);

    // bilinear gather + weighted sum: one wave per (b,n,head)
    deform_sample<<<(MROWS * NHEAD) / 8, 256, 0, stream>>>(proj, offatt, ref, mix_bf);

    // output projection: [32768,256] x [256,256]
    gemm_bf16_wmma<<<dim3(256 / BN, MROWS / BM), 256, 0, stream>>>(
        mix_bf, Wt_out, b_out, out, MROWS, 256, 256);
}
